// RandomBatchTripletLoss_86612310492001
// MI455X (gfx1250) — matrix operand order repaired and checked by reading, not compiled, not run on hardware
//
#include <hip/hip_runtime.h>
#include <math.h>

typedef __attribute__((ext_vector_type(16))) __bf16 v16bf;
typedef __attribute__((ext_vector_type(8)))  float  v8f;

#define WMMA_BF16(A, B, C) \
  __builtin_amdgcn_wmma_f32_16x16x32_bf16(false, (A), false, (B), (short)0, (C), false, false)

union FragU { uint4 q[2]; v16bf v; };

constexpr int NB = 4096;   // batch
constexpr int ND = 512;    // feature dim

// workspace layout (units: floats)
constexpr size_t OFF_SQ    = 0;                         // [4096]
constexpr size_t OFF_PSC   = 4096;                      // [4096] pscore
constexpr size_t OFF_SPOS  = 8192;                      // [4096] score_pos
constexpr size_t OFF_SPOS1 = 12288;                     // [4096] score_pos1
constexpr size_t OFF_FB    = 16384;                     // [4096] dist_neg[i,0] fallback
constexpr size_t OFF_PMIN  = 20480;                     // [8][4096] partial min
constexpr size_t OFF_PLOG  = OFF_PMIN + 8 * 4096;       // [8][4096] partial best logit
constexpr size_t OFF_PD    = OFF_PLOG + 8 * 4096;       // [8][4096] partial best dist
constexpr size_t OFF_ACT   = OFF_PD  + 8 * 4096;        // [4096] active flags
constexpr size_t OFF_ACC   = OFF_ACT + 4096;            // [4096] accuracy flags
constexpr size_t OFF_FHI   = OFF_ACC + 4096;            // bf16[4096*512] as ushort
constexpr size_t OFF_FLO   = OFF_FHI + (size_t)NB * ND / 2;

__device__ __forceinline__ float gumbelf(float u) {
  return -logf(-logf(u + 1e-12f) + 1e-12f);
}

__device__ __forceinline__ unsigned short bf16_rn(float f) {
  unsigned int u = __float_as_uint(f);
  unsigned int r = (u + 0x7FFFu + ((u >> 16) & 1u)) >> 16;
  return (unsigned short)r;
}

__device__ __forceinline__ void split_bf16(float f, unsigned short& h, unsigned short& l) {
  h = bf16_rn(f);
  float hf = __uint_as_float(((unsigned int)h) << 16);
  l = bf16_rn(f - hf);
}

// ---------------------------------------------------------------------------
// Kernel 0: row squared-norms + bf16 hi/lo split of features.
// One wave per row; lane handles 16 contiguous floats.
// ---------------------------------------------------------------------------
__global__ __launch_bounds__(256) void prep_kernel(
    const float* __restrict__ F, float* __restrict__ sq,
    unsigned short* __restrict__ fhi, unsigned short* __restrict__ flo) {
  int wave = threadIdx.x >> 5, lane = threadIdx.x & 31;
  int row  = blockIdx.x * 8 + wave;
  const float4* F4 = reinterpret_cast<const float4*>(F + (size_t)row * ND);
  ushort4* H4 = reinterpret_cast<ushort4*>(fhi + (size_t)row * ND);
  ushort4* L4 = reinterpret_cast<ushort4*>(flo + (size_t)row * ND);
  float s = 0.0f;
#pragma unroll
  for (int t = 0; t < 4; ++t) {
    float4 v = F4[lane * 4 + t];
    s += v.x * v.x + v.y * v.y + v.z * v.z + v.w * v.w;
    ushort4 h, l;
    split_bf16(v.x, h.x, l.x); split_bf16(v.y, h.y, l.y);
    split_bf16(v.z, h.z, l.z); split_bf16(v.w, h.w, l.w);
    H4[lane * 4 + t] = h;
    L4[lane * 4 + t] = l;
  }
#pragma unroll
  for (int off = 16; off >= 1; off >>= 1) s += __shfl_xor(s, off, 32);
  if (lane == 0) sq[row] = s;
}

// ---------------------------------------------------------------------------
// Kernel 1: positive-side stats per class block (8 rows). 64 threads / block.
// Thread t: pair (i = 8c + t/8, j = 8c + t%8), fp32 dot from LDS.
// ---------------------------------------------------------------------------
__global__ __launch_bounds__(64) void pos_kernel(
    const float* __restrict__ F, const float* __restrict__ sq,
    const float* __restrict__ u_pos, float* __restrict__ pscore,
    float* __restrict__ spos, float* __restrict__ spos1) {
  __shared__ float rows[8][516];
  int c = blockIdx.x;
  int t = threadIdx.x;
  for (int idx = t; idx < 8 * ND; idx += 64) {
    int r = idx >> 9, k = idx & 511;
    rows[r][k] = F[(size_t)(c * 8 + r) * ND + k];
  }
  __syncthreads();
  int r = t >> 3, q = t & 7;
  int i = c * 8 + r, j = c * 8 + q;
  float dot = 0.0f;
#pragma unroll 8
  for (int k = 0; k < ND; ++k) dot = fmaf(rows[r][k], rows[q][k], dot);
  float d = fmaxf(sq[i] + sq[j] - 2.0f * dot, 1e-8f);
  bool valid = (q != r);
  float dp = valid ? d : 0.0f;                    // dist_pos entry
  float lg = valid ? (logf(dp + 1e-30f) + gumbelf(u_pos[(size_t)i * NB + j]))
                   : -1e30f;
  float mx = dp, bd = dp, bl = lg;
#pragma unroll
  for (int off = 1; off < 8; off <<= 1) {
    mx = fmaxf(mx, __shfl_xor(mx, off, 32));
    float ol = __shfl_xor(bl, off, 32);
    float od = __shfl_xor(bd, off, 32);
    if (ol > bl) { bl = ol; bd = od; }
  }
  if (q == 0) {
    pscore[i] = bd;
    spos[i]   = sqrtf(bd);
    spos1[i]  = sqrtf(mx);
  }
}

// ---------------------------------------------------------------------------
// Kernel 2: main Gram + negative reductions.
// Grid (64,4): 64-row strip x 1024-col segment per WG, 256 threads (8 waves).
// bf16x3 split GEMM with v_wmma_f32_16x16x32_bf16; A strip (hi+lo) staged in
// 133 KB LDS (pitch 65 uint4 for bank skew); B streamed from L2.
// ---------------------------------------------------------------------------
__device__ __forceinline__ void neg_epilogue(
    const v8f& acc, int cb, int ln, int half, int rowTile,
    const float (&sqv)[8], const float (&pscv)[8],
    const float* __restrict__ sq, const float* __restrict__ u_neg,
    float* __restrict__ fb, float (&mn)[8], float (&blg)[8], float (&bd)[8]) {
  int j = cb + ln;
  float sqj = sq[j];
  int jb = j >> 3;
  const float* ur = u_neg + (size_t)(rowTile + 8 * half) * NB + j;
#pragma unroll
  for (int r = 0; r < 8; ++r) {
    int i = rowTile + r + 8 * half;
    float d  = fmaxf(sqv[r] + sqj - 2.0f * acc[r], 1e-8f);
    float dn = ((i >> 3) == jb) ? 1e25f : d;      // dist_neg (same-class -> 1e25)
    mn[r] = fminf(mn[r], dn);
    float g  = gumbelf(ur[(size_t)r * NB]);
    float lg = g - dn;
    if ((dn < pscv[r]) && (lg > blg[r])) { blg[r] = lg; bd[r] = dn; }
    if (j == 0) fb[i] = dn;                       // fallback index is always 0
  }
}

__global__ __launch_bounds__(256) void neg_kernel(
    const unsigned short* __restrict__ fhi, const unsigned short* __restrict__ flo,
    const float* __restrict__ sq, const float* __restrict__ pscore,
    const float* __restrict__ u_neg, float* __restrict__ fb,
    float* __restrict__ pmin, float* __restrict__ plog, float* __restrict__ pd) {
  extern __shared__ char smem[];
  uint4* sAh = reinterpret_cast<uint4*>(smem);                 // 64 rows * 65 uint4
  uint4* sAl = reinterpret_cast<uint4*>(smem + 64 * 65 * 16);
  const int rowBase = blockIdx.x * 64;
  const int ns      = blockIdx.y;
  const int colSeg  = ns * 1024;
  const int tid     = threadIdx.x;

  const uint4* Fh4 = reinterpret_cast<const uint4*>(fhi);
  const uint4* Fl4 = reinterpret_cast<const uint4*>(flo);

  // stage A strip (64 rows x 512 bf16, hi+lo) into LDS
  for (int idx = tid; idx < 64 * 64; idx += 256) {
    int r = idx >> 6, cq = idx & 63;
    sAh[r * 65 + cq] = Fh4[(size_t)(rowBase + r) * 64 + cq];
    sAl[r * 65 + cq] = Fl4[(size_t)(rowBase + r) * 64 + cq];
  }
  __syncthreads();

  const int w = tid >> 5, lane = tid & 31;
  const int half = lane >> 4, ln = lane & 15;
  const int rw = w & 3, cwofs = (w >> 2) * 16;
  const int rowTile = rowBase + 16 * rw;

  float sqv[8], pscv[8];
#pragma unroll
  for (int r = 0; r < 8; ++r) {
    int i = rowTile + r + 8 * half;
    sqv[r]  = sq[i];
    pscv[r] = pscore[i] + 1e-4f;
  }
  float mn[8], blg[8], bd[8];
#pragma unroll
  for (int r = 0; r < 8; ++r) { mn[r] = 3e38f; blg[r] = -3e38f; bd[r] = 1e25f; }

  const uint4* aH = sAh + (16 * rw + ln) * 65;
  const uint4* aL = sAl + (16 * rw + ln) * 65;

  for (int it = 0; it < 16; ++it) {
    int c0 = colSeg + it * 64 + cwofs;
    int c1 = c0 + 32;
    v8f acc0 = {};
    v8f acc1 = {};
    size_t b0 = (size_t)(c0 + ln) * 64 + 2 * half;
    size_t b1 = (size_t)(c1 + ln) * 64 + 2 * half;
#pragma unroll 2
    for (int k8 = 0; k8 < 16; ++k8) {
      FragU ah, al, bh0, bl0, bh1, bl1;
      int ai = k8 * 4 + 2 * half;
      ah.q[0] = aH[ai]; ah.q[1] = aH[ai + 1];
      al.q[0] = aL[ai]; al.q[1] = aL[ai + 1];
      size_t bi = (size_t)k8 * 4;
      bh0.q[0] = Fh4[b0 + bi]; bh0.q[1] = Fh4[b0 + bi + 1];
      bl0.q[0] = Fl4[b0 + bi]; bl0.q[1] = Fl4[b0 + bi + 1];
      bh1.q[0] = Fh4[b1 + bi]; bh1.q[1] = Fh4[b1 + bi + 1];
      bl1.q[0] = Fl4[b1 + bi]; bl1.q[1] = Fl4[b1 + bi + 1];
      acc0 = WMMA_BF16(ah.v, bh0.v, acc0);
      acc1 = WMMA_BF16(ah.v, bh1.v, acc1);
      acc0 = WMMA_BF16(al.v, bh0.v, acc0);
      acc1 = WMMA_BF16(al.v, bh1.v, acc1);
      acc0 = WMMA_BF16(ah.v, bl0.v, acc0);
      acc1 = WMMA_BF16(ah.v, bl1.v, acc1);
    }
    neg_epilogue(acc0, c0, ln, half, rowTile, sqv, pscv, sq, u_neg, fb, mn, blg, bd);
    neg_epilogue(acc1, c1, ln, half, rowTile, sqv, pscv, sq, u_neg, fb, mn, blg, bd);
  }

  // cross-lane reduce within each 16-lane half (row-wise over 16 columns)
#pragma unroll
  for (int off = 1; off < 16; off <<= 1) {
#pragma unroll
    for (int r = 0; r < 8; ++r) {
      mn[r] = fminf(mn[r], __shfl_xor(mn[r], off, 32));
      float ol = __shfl_xor(blg[r], off, 32);
      float od = __shfl_xor(bd[r],  off, 32);
      if (ol > blg[r]) { blg[r] = ol; bd[r] = od; }
    }
  }
  if (ln == 0) {
    int p = ns * 2 + (w >> 2);
#pragma unroll
    for (int r = 0; r < 8; ++r) {
      int i = rowTile + r + 8 * half;
      pmin[(size_t)p * NB + i] = mn[r];
      plog[(size_t)p * NB + i] = blg[r];
      pd[(size_t)p * NB + i]   = bd[r];
    }
  }
}

// ---------------------------------------------------------------------------
// Kernel 3: per-row finalize (combine 8 partials, compute loss/flags).
// ---------------------------------------------------------------------------
__global__ __launch_bounds__(256) void finalize_kernel(
    const float* __restrict__ pmin, const float* __restrict__ plog,
    const float* __restrict__ pd, const float* __restrict__ fb,
    const float* __restrict__ spos, const float* __restrict__ spos1,
    float* __restrict__ act, float* __restrict__ accf, float* __restrict__ out) {
  int i = blockIdx.x * 256 + threadIdx.x;
  float mnv = 3e38f, bl = -3e38f, bdv = 1e25f;
#pragma unroll
  for (int p = 0; p < 8; ++p) {
    mnv = fminf(mnv, pmin[(size_t)p * NB + i]);
    float l = plog[(size_t)p * NB + i];
    if (l > bl) { bl = l; bdv = pd[(size_t)p * NB + i]; }
  }
  float sneg1 = sqrtf(mnv);
  float dn    = (bl > -1e29f) ? bdv : fb[i];      // any(mask2) ? sampled : fallback(j=0)
  float sneg  = sqrtf(dn);
  float diff  = 1e-4f + spos[i] - sneg;
  float loss  = fmaxf(diff, 0.0f) + log1pf(expf(-fabsf(diff)));  // softplus
  out[i]  = loss;
  act[i]  = (diff > 0.0f) ? 1.0f : 0.0f;
  accf[i] = (spos1[i] < sneg1) ? 1.0f : 0.0f;
}

// ---------------------------------------------------------------------------
// Kernel 4: scalar reduction for avg_active and accuracy.
// ---------------------------------------------------------------------------
__global__ __launch_bounds__(256) void reduce_kernel(
    const float* __restrict__ act, const float* __restrict__ accf,
    float* __restrict__ out) {
  __shared__ float sA[256], sB[256];
  int t = threadIdx.x;
  float a = 0.0f, b = 0.0f;
  for (int i = t; i < NB; i += 256) { a += act[i]; b += accf[i]; }
  sA[t] = a; sB[t] = b;
  __syncthreads();
  for (int s = 128; s > 0; s >>= 1) {
    if (t < s) { sA[t] += sA[t + s]; sB[t] += sB[t + s]; }
    __syncthreads();
  }
  if (t == 0) {
    out[NB]     = sA[0] / (float)NB;           // avg_active
    out[NB + 1] = 100.0f * sB[0] / (float)NB;  // accuracy
  }
}

extern "C" void kernel_launch(void* const* d_in, const int* in_sizes, int n_in,
                              void* d_out, int out_size, void* d_ws, size_t ws_size,
                              hipStream_t stream) {
  (void)in_sizes; (void)n_in; (void)out_size; (void)ws_size;
  const float* F     = (const float*)d_in[0];
  const float* u_pos = (const float*)d_in[1];
  const float* u_neg = (const float*)d_in[2];
  float* out = (float*)d_out;
  float* ws  = (float*)d_ws;

  float* sq     = ws + OFF_SQ;
  float* pscore = ws + OFF_PSC;
  float* spos   = ws + OFF_SPOS;
  float* spos1  = ws + OFF_SPOS1;
  float* fb     = ws + OFF_FB;
  float* pmin   = ws + OFF_PMIN;
  float* plog   = ws + OFF_PLOG;
  float* pd     = ws + OFF_PD;
  float* act    = ws + OFF_ACT;
  float* accf   = ws + OFF_ACC;
  unsigned short* fhi = (unsigned short*)(ws + OFF_FHI);
  unsigned short* flo = (unsigned short*)(ws + OFF_FLO);

  prep_kernel<<<NB / 8, 256, 0, stream>>>(F, sq, fhi, flo);
  pos_kernel<<<NB / 8, 64, 0, stream>>>(F, sq, u_pos, pscore, spos, spos1);

  dim3 g2(64, 4, 1);
  size_t shmem = 2u * 64u * 65u * 16u;  // 133,120 B (A hi+lo, padded)
  neg_kernel<<<g2, 256, shmem, stream>>>(fhi, flo, sq, pscore, u_neg, fb, pmin, plog, pd);

  finalize_kernel<<<NB / 256, 256, 0, stream>>>(pmin, plog, pd, fb, spos, spos1, act, accf, out);
  reduce_kernel<<<1, 256, 0, stream>>>(act, accf, out);
}
